// BehavioralRotaryAttentionV12_84748294684700
// MI455X (gfx1250) — compile-verified, matrix operand order checked by reading, not run on hardware
//
#include <hip/hip_runtime.h>
#include <hip/hip_bf16.h>

// ===========================================================================
// BehavioralRotaryAttentionV12 for MI455X (gfx1250, wave32, WMMA).
//
// ~69 GFLOP over a ~70MB L2-resident working set -> compute-bound.
// bf16 inputs + f32-accumulate WMMA (v_wmma_f32_16x16x32_bf16) for all GEMMs;
// GLOBAL_LOAD_ASYNC_TO_LDS_B128 (ASYNCcnt) double-buffered tile staging in the
// projection GEMMs; fp32 for softmax stats / rotary / bias / residual / LN.
// All fragment gathers are 2x b128 per lane (ISA 7.12.2 16-bit layout puts a
// lane's K-pairs in two contiguous 16B runs).
// ===========================================================================

#define DI __device__ __forceinline__

typedef __attribute__((ext_vector_type(16))) __bf16 v16bf;
typedef __attribute__((ext_vector_type(8)))  float  v8f;
typedef __attribute__((ext_vector_type(8)))  int    v8i;

static constexpr int Bb  = 2;
static constexpr int Ls  = 2048;
static constexpr int Ds  = 1024;
static constexpr int Hs  = 16;
static constexpr int HDs = 64;
static constexpr int NT  = Bb * Ls;   // 4096 tokens

// round-to-nearest-even fp32 -> bf16 (raw bits)
DI unsigned short f2bf_raw(float f) {
  unsigned int u = __builtin_bit_cast(unsigned int, f);
  unsigned int r = (u + 0x7fffu + ((u >> 16) & 1u)) >> 16;
  return (unsigned short)r;
}

DI v8f vzero8() { v8f z = {0.f,0.f,0.f,0.f,0.f,0.f,0.f,0.f}; return z; }

// Build a 16x32 bf16 A/B fragment for this lane from a row pointer.
// ISA 7.12.2: lanes 0-15 hold K={0..7,16..23}, lanes 16-31 K={8..15,24..31}
// (khalf = 8*(lane>=16)), i.e. two contiguous 16-byte runs -> two b128 loads.
DI v16bf frag_from_row(const unsigned short* p, int khalf) {
  uint4 a = *(const uint4*)(p + khalf);        // elements khalf .. khalf+7
  uint4 b = *(const uint4*)(p + 16 + khalf);   // elements 16+khalf .. +7
  v8i v;
  v[0]=(int)a.x; v[1]=(int)a.y; v[2]=(int)a.z; v[3]=(int)a.w;
  v[4]=(int)b.x; v[5]=(int)b.y; v[6]=(int)b.z; v[7]=(int)b.w;
  return __builtin_bit_cast(v16bf, v);
}

DI v8f wmma_bf16(v16bf a, v16bf b, v8f c) {
  // D = A(16x32 bf16) x B(32x16 bf16) + C(16x16 f32)
  return __builtin_amdgcn_wmma_f32_16x16x32_bf16(
      /*neg_a=*/false, a, /*neg_b=*/false, b,
      /*c_mod=*/(short)0, c, /*reuse_a=*/false, /*reuse_b=*/false);
}

// --- CDNA5 async global->LDS copy (ASYNCcnt path), GV addressing ------------
// LDS address = low 32 bits of the generic shared-space address (ISA 10.2:
// LDS aperture truncates to addr[31:0]).
DI unsigned lds_addr32(const void* p) {
  return (unsigned)(unsigned long long)p;
}
DI void async_ld_b128(unsigned ldsaddr, const void* gaddr) {
  asm volatile("global_load_async_to_lds_b128 %0, %1, off"
               :: "v"(ldsaddr), "v"(gaddr) : "memory");
}
DI void wait_async0() {
  asm volatile("s_wait_asynccnt 0x0" ::: "memory");
}

// ---------------------------------------------------------------------------
// fp32 -> bf16 elementwise conversion
// ---------------------------------------------------------------------------
__global__ __launch_bounds__(256) void f32_to_bf16_kernel(
    const float* __restrict__ src, unsigned short* __restrict__ dst, int n) {
  int i = blockIdx.x * 256 + threadIdx.x;
  if (i < n) dst[i] = f2bf_raw(src[i]);
}

// ---------------------------------------------------------------------------
// Tiled WMMA GEMM: out[m][n] = sum_k A[m][k] * W[n][k]  (i.e. A @ W^T)
// Tile: 128(M) x 64(N), BK=32, 8 waves; wave w owns rows 16w..16w+15.
// Double-buffered LDS staging via GLOBAL_LOAD_ASYNC_TO_LDS_B128.
// N-tiles are 64-wide and head-aligned -> rotary (d vs d^2-tile) is wave-local.
// MODE 0: Q (bias+rotary, write bf16 [b,h,l,d])
// MODE 1: K (bias+rotary, write bf16 [b,h,l,d])
// MODE 2: V (bias,        write bf16 [b,h,d,l])  (transposed for PV B-frags)
// MODE 3: O (bias+residual, write f32 [token][D])
// ---------------------------------------------------------------------------
template <int MODE>
__global__ __launch_bounds__(256) void gemm_wmma_kernel(
    const unsigned short* __restrict__ A,   // bf16 [NT][D]
    const unsigned short* __restrict__ W,   // bf16 [D][D] (row n, col k)
    const float* __restrict__ bias,         // [D]
    const float* __restrict__ phi,          // [B,L,H]  (modes 0,1)
    const float* __restrict__ resid,        // f32 [NT][D] (mode 3)
    unsigned short* __restrict__ outb,      // bf16 out (modes 0,1,2)
    float* __restrict__ outf) {             // f32 out (mode 3)
  // Unpadded 32-element K stride: rows are 64B so all 16B chunks are aligned
  // for both the async b128 stores and the b128 fragment gathers.
  __shared__ unsigned short As[2][128][32];
  __shared__ unsigned short Bs[2][64][32];

  const int n0 = blockIdx.x * 64;
  const int m0 = blockIdx.y * 128;
  const int tid  = threadIdx.x;
  const int lane = tid & 31;
  const int wv   = tid >> 5;       // 0..7
  const int hi   = lane >> 4;      // 0/1
  const int ln   = lane & 15;
  const int khalf = hi * 8;

  // Stage one 128x32 A tile (512 b128 chunks) + 64x32 W tile (256 chunks).
  auto stage = [&](int buf, int k0) {
#pragma unroll
    for (int i = 0; i < 2; ++i) {
      int c = tid + 256 * i;            // 0..511
      int row = c >> 2, seg = (c & 3) * 8;
      async_ld_b128(lds_addr32(&As[buf][row][seg]),
                    A + (m0 + row) * Ds + k0 + seg);
    }
    {
      int row = tid >> 2, seg = (tid & 3) * 8;
      async_ld_b128(lds_addr32(&Bs[buf][row][seg]),
                    W + (n0 + row) * Ds + k0 + seg);
    }
  };

  v8f acc[4] = {vzero8(), vzero8(), vzero8(), vzero8()};

  stage(0, 0);
  for (int k0 = 0, it = 0; k0 < Ds; k0 += 32, ++it) {
    const int buf = it & 1;
    wait_async0();        // my wave's async copies for `buf` have landed
    __syncthreads();      // everyone's copies have landed
    if (k0 + 32 < Ds) stage(buf ^ 1, k0 + 32);  // prefetch next tile

    v16bf af = frag_from_row(&As[buf][16 * wv + ln][0], khalf);
#pragma unroll
    for (int t = 0; t < 4; ++t) {
      v16bf bf_ = frag_from_row(&Bs[buf][t * 16 + ln][0], khalf);
      acc[t] = wmma_bf16(af, bf_, acc[t]);
    }
    // No trailing barrier: next iteration's wait+barrier precedes any
    // overwrite of buf^1, and buf^1's readers finished before this barrier.
  }

  // ---------------- epilogue ----------------
  // C layout: lane holds col n = n0+16t+ln, rows m = m0+16*wv + r + 8*hi.
#pragma unroll
  for (int t = 0; t < 4; ++t) {
    float bv = bias[n0 + 16 * t + ln];
#pragma unroll
    for (int r = 0; r < 8; ++r) acc[t][r] += bv;
  }

  const int mrow0 = m0 + 16 * wv + 8 * hi;

  if (MODE == 0 || MODE == 1) {
    const int h = n0 >> 6;  // 64-wide N tile == one head
    float cs[8], sn[8];
#pragma unroll
    for (int r = 0; r < 8; ++r) {
      int tok = mrow0 + r;                       // == b*L + pos
      float ph = phi[tok * Hs + h];
      __sincosf(ph, &sn[r], &cs[r]);
    }
#pragma unroll
    for (int t = 0; t < 4; ++t) {
      const int d = 16 * t + ln;
      const float sign = (t < 2) ? -1.f : 1.f;   // rotate_half: [-x2, x1]
#pragma unroll
      for (int r = 0; r < 8; ++r) {
        float o = acc[t][r] * cs[r] + sign * acc[t ^ 2][r] * sn[r];
        int tok = mrow0 + r;
        int b = tok >> 11, pos = tok & 2047;
        outb[((b * Hs + h) * Ls + pos) * HDs + d] = f2bf_raw(o);
      }
    }
  } else if (MODE == 2) {
    const int h = n0 >> 6;
#pragma unroll
    for (int t = 0; t < 4; ++t) {
      const int d = 16 * t + ln;
#pragma unroll
      for (int r = 0; r < 8; ++r) {
        int tok = mrow0 + r;
        int b = tok >> 11, pos = tok & 2047;
        outb[((b * Hs + h) * HDs + d) * Ls + pos] = f2bf_raw(acc[t][r]);
      }
    }
  } else {  // MODE 3: O-proj + residual
#pragma unroll
    for (int t = 0; t < 4; ++t) {
      const int n = n0 + 16 * t + ln;
#pragma unroll
      for (int r = 0; r < 8; ++r) {
        int tok = mrow0 + r;
        outf[tok * Ds + n] = acc[t][r] + resid[tok * Ds + n];
      }
    }
  }
}

// ---------------------------------------------------------------------------
// Flash attention with rotary sync-mask.
// Grid: (q_tile 0..31, bh 0..31). Block: 128 threads = 4 waves.
// Wave owns 16 query rows; streams 32 keys/iter: 4 WMMA (scores) + 4 (PV).
// ---------------------------------------------------------------------------
__global__ __launch_bounds__(128) void attn_kernel(
    const unsigned short* __restrict__ qr,   // bf16 [b,h,l,d]
    const unsigned short* __restrict__ kr,   // bf16 [b,h,l,d]
    const unsigned short* __restrict__ vt,   // bf16 [b,h,d,l]
    const float* __restrict__ phi,           // [B,L,H]
    const float* __restrict__ amask,         // [B,1,1,L]
    unsigned short* __restrict__ ctxb) {     // bf16 [token][D]
  __shared__ unsigned short Pt[4][16][32];   // per-wave P tile (16x32)

  const int bh = blockIdx.y;
  const int b = bh >> 4, h = bh & 15;
  const int wv   = threadIdx.x >> 5;
  const int lane = threadIdx.x & 31;
  const int hi = lane >> 4, ln = lane & 15;
  const int khalf = hi * 8;
  const int qbase = blockIdx.x * 64 + wv * 16;

  const unsigned short* qbh = qr + (size_t)bh * Ls * HDs;
  const unsigned short* kbh = kr + (size_t)bh * Ls * HDs;
  const unsigned short* vbh = vt + (size_t)bh * HDs * Ls;

  // Q A-fragments (contraction over HD=64 -> two k-steps)
  v16bf aQ[2];
  {
    const int qrow = qbase + ln;
#pragma unroll
    for (int kk = 0; kk < 2; ++kk)
      aQ[kk] = frag_from_row(qbh + qrow * HDs + kk * 32, khalf);
  }

  // per-lane query phases for my 8 C-rows
  float phq[8];
#pragma unroll
  for (int r = 0; r < 8; ++r)
    phq[r] = phi[(b * Ls + qbase + r + 8 * hi) * Hs + h];

  float rowm[8], rowl[8];
#pragma unroll
  for (int r = 0; r < 8; ++r) { rowm[r] = -3.0e38f; rowl[r] = 0.f; }
  v8f cacc[4] = {vzero8(), vzero8(), vzero8(), vzero8()};

  for (int kb = 0; kb < Ls; kb += 32) {
    // ---- scores for 32 keys: two 16-wide N tiles ----
    v8f s[2] = {vzero8(), vzero8()};
#pragma unroll
    for (int nt = 0; nt < 2; ++nt) {
      const int key = kb + nt * 16 + ln;
#pragma unroll
      for (int kk = 0; kk < 2; ++kk) {
        v16bf bK = frag_from_row(kbh + key * HDs + kk * 32, khalf);
        s[nt] = wmma_bf16(aQ[kk], bK, s[nt]);
      }
    }

    // ---- scale + attention_mask + data-dependent sync mask ----
    float pk[2], am[2];
#pragma unroll
    for (int nt = 0; nt < 2; ++nt) {
      int key = kb + nt * 16 + ln;
      pk[nt] = phi[(b * Ls + key) * Hs + h];
      am[nt] = amask[b * Ls + key];
    }
#pragma unroll
    for (int nt = 0; nt < 2; ++nt) {
      const int key = kb + nt * 16 + ln;
#pragma unroll
      for (int r = 0; r < 8; ++r) {
        float sc = s[nt][r] * 0.125f + am[nt];
        int m = qbase + r + 8 * hi;
        if (__cosf(phq[r] - pk[nt]) < -0.7f && m != key) sc = -1e9f;
        s[nt][r] = sc;
      }
    }
    // NOTE: reference's "all rows masked -> attend to self" edge case cannot
    // trigger here (diagonal is never sync-masked and attention_mask == 0).

    // ---- online softmax (rows live across 16-lane groups; xor masks 1..8
    //      stay within a group, matching the wave32 C layout) ----
#pragma unroll
    for (int r = 0; r < 8; ++r) {
      float mx = fmaxf(s[0][r], s[1][r]);
#pragma unroll
      for (int off = 1; off < 16; off <<= 1) mx = fmaxf(mx, __shfl_xor(mx, off, 32));
      float mnew = fmaxf(rowm[r], mx);
      float corr = __expf(rowm[r] - mnew);
      float p0 = __expf(s[0][r] - mnew);
      float p1 = __expf(s[1][r] - mnew);
      float rs = p0 + p1;
#pragma unroll
      for (int off = 1; off < 16; off <<= 1) rs += __shfl_xor(rs, off, 32);
      rowl[r] = rowl[r] * corr + rs;
      rowm[r] = mnew;
#pragma unroll
      for (int t = 0; t < 4; ++t) cacc[t][r] *= corr;
      Pt[wv][r + 8 * hi][ln]      = f2bf_raw(p0);
      Pt[wv][r + 8 * hi][16 + ln] = f2bf_raw(p1);
    }
    __syncthreads();  // uniform trip count across waves; orders LDS P tile

    // ---- P C-layout -> A-fragment via LDS (two ds_load_b128) ----
    v16bf pA = frag_from_row(&Pt[wv][ln][0], khalf);

    // ---- ctx += P(16x32) @ V(32x64): 4 d-tiles; V^T layout -> contiguous K
#pragma unroll
    for (int t = 0; t < 4; ++t) {
      const int d = 16 * t + ln;
      v16bf bV = frag_from_row(vbh + d * Ls + kb, khalf);
      cacc[t] = wmma_bf16(pA, bV, cacc[t]);
    }
    __syncthreads();
  }

  // ---- normalize & write ctx as bf16 [token][D] ----
  float inv[8];
#pragma unroll
  for (int r = 0; r < 8; ++r) inv[r] = 1.f / (rowl[r] + 1e-30f);
#pragma unroll
  for (int t = 0; t < 4; ++t) {
    const int col = h * HDs + 16 * t + ln;
#pragma unroll
    for (int r = 0; r < 8; ++r) {
      int tok = b * Ls + qbase + r + 8 * hi;
      ctxb[tok * Ds + col] = f2bf_raw(cacc[t][r] * inv[r]);
    }
  }
}

// ---------------------------------------------------------------------------
// LayerNorm over last dim (1024), block per token row.
// ---------------------------------------------------------------------------
__global__ __launch_bounds__(256) void ln_kernel(
    const float* __restrict__ x, const float* __restrict__ g,
    const float* __restrict__ bta, float* __restrict__ out) {
  __shared__ float ss[8], ss2[8];
  const int row = blockIdx.x;
  const float* xr = x + row * Ds;
  float s = 0.f, s2 = 0.f;
  for (int i = threadIdx.x; i < Ds; i += 256) {
    float v = xr[i];
    s += v; s2 += v * v;
  }
#pragma unroll
  for (int off = 1; off < 32; off <<= 1) {
    s  += __shfl_xor(s,  off, 32);
    s2 += __shfl_xor(s2, off, 32);
  }
  int wv = threadIdx.x >> 5, lane = threadIdx.x & 31;
  if (lane == 0) { ss[wv] = s; ss2[wv] = s2; }
  __syncthreads();
  if (wv == 0) {
    s  = (lane < 8) ? ss[lane]  : 0.f;
    s2 = (lane < 8) ? ss2[lane] : 0.f;
#pragma unroll
    for (int off = 1; off < 8; off <<= 1) {
      s  += __shfl_xor(s,  off, 32);
      s2 += __shfl_xor(s2, off, 32);
    }
    if (lane == 0) { ss[0] = s; ss2[0] = s2; }
  }
  __syncthreads();
  float mean = ss[0] * (1.f / Ds);
  float var  = ss2[0] * (1.f / Ds) - mean * mean;
  float rinv = rsqrtf(var + 1e-12f);
  for (int i = threadIdx.x; i < Ds; i += 256)
    out[row * Ds + i] = (xr[i] - mean) * rinv * g[i] + bta[i];
}

// ---------------------------------------------------------------------------
// Launch. Workspace layout (64 MiB total):
//   hsb 8MiB | Wq/Wk/Wv/Wo bf16 2MiB ea | qr 8 | kr 8 | vt 8 | ctx 8 | outf 16
// ---------------------------------------------------------------------------
extern "C" void kernel_launch(void* const* d_in, const int* in_sizes, int n_in,
                              void* d_out, int out_size, void* d_ws, size_t ws_size,
                              hipStream_t stream) {
  (void)in_sizes; (void)n_in; (void)out_size; (void)ws_size;
  const float* hidden = (const float*)d_in[0];
  const float* amask  = (const float*)d_in[1];
  const float* phi    = (const float*)d_in[2];
  const float* Wq = (const float*)d_in[3];  const float* bq = (const float*)d_in[4];
  const float* Wk = (const float*)d_in[5];  const float* bk = (const float*)d_in[6];
  const float* Wv = (const float*)d_in[7];  const float* bv = (const float*)d_in[8];
  const float* Wo = (const float*)d_in[9];  const float* bo = (const float*)d_in[10];
  const float* lng = (const float*)d_in[11];
  const float* lnb = (const float*)d_in[12];

  char* ws = (char*)d_ws;
  const size_t MiB = 1ull << 20;
  unsigned short* hsb  = (unsigned short*)(ws + 0 * MiB);
  unsigned short* wqb  = (unsigned short*)(ws + 8 * MiB);
  unsigned short* wkb  = (unsigned short*)(ws + 10 * MiB);
  unsigned short* wvb  = (unsigned short*)(ws + 12 * MiB);
  unsigned short* wob  = (unsigned short*)(ws + 14 * MiB);
  unsigned short* qrp  = (unsigned short*)(ws + 16 * MiB);
  unsigned short* krp  = (unsigned short*)(ws + 24 * MiB);
  unsigned short* vtp  = (unsigned short*)(ws + 32 * MiB);
  unsigned short* ctxb = (unsigned short*)(ws + 40 * MiB);
  float*          outf = (float*)(ws + 48 * MiB);

  const int nHS = NT * Ds;      // 4194304
  const int nW  = Ds * Ds;      // 1048576
  f32_to_bf16_kernel<<<(nHS + 255) / 256, 256, 0, stream>>>(hidden, hsb, nHS);
  f32_to_bf16_kernel<<<(nW  + 255) / 256, 256, 0, stream>>>(Wq, wqb, nW);
  f32_to_bf16_kernel<<<(nW  + 255) / 256, 256, 0, stream>>>(Wk, wkb, nW);
  f32_to_bf16_kernel<<<(nW  + 255) / 256, 256, 0, stream>>>(Wv, wvb, nW);
  f32_to_bf16_kernel<<<(nW  + 255) / 256, 256, 0, stream>>>(Wo, wob, nW);

  dim3 gg(Ds / 64, NT / 128);   // (16, 32)
  gemm_wmma_kernel<0><<<gg, 256, 0, stream>>>(hsb, wqb, bq, phi, nullptr, qrp, nullptr);
  gemm_wmma_kernel<1><<<gg, 256, 0, stream>>>(hsb, wkb, bk, phi, nullptr, krp, nullptr);
  gemm_wmma_kernel<2><<<gg, 256, 0, stream>>>(hsb, wvb, bv, phi, nullptr, vtp, nullptr);

  attn_kernel<<<dim3(Ls / 64, Bb * Hs), 128, 0, stream>>>(qrp, krp, vtp, phi, amask, ctxb);

  gemm_wmma_kernel<3><<<gg, 256, 0, stream>>>(ctxb, wob, bo, phi, hidden, nullptr, outf);

  ln_kernel<<<NT, 256, 0, stream>>>(outf, lng, lnb, (float*)d_out);
}
